// CGAT_35871566856658
// MI455X (gfx1250) — matrix operand order repaired and checked by reading, not compiled, not run on hardware
//
#include <hip/hip_runtime.h>

typedef __attribute__((ext_vector_type(16))) _Float16 v16h;
typedef __attribute__((ext_vector_type(8)))  float    v8f;

#define ALPHA 0.2f
constexpr int B_ = 2, V_ = 8192, T_ = 8, F0_ = 32, F1_ = 32;
constexpr int COLS = T_ * F1_;   // 256 flattened (t,g) columns

__device__ __forceinline__ float lrelu(float x) { return fmaxf(x, ALPHA * x); }

// ---------------------------------------------------------------------------
// Kernel 1: h[b,v,t,g] = lrelu(x[b,v,t,:]·W[g,:] + b[g])
//   stores h as f32 (row major, for exact ht reduction) and as f16 transposed
//   Ht[b][col][j] so WMMA B-fragments are contiguous 16B loads.
// One wave per (b,v,t) row; lane = g.  x row broadcast via __shfl.
// ---------------------------------------------------------------------------
__global__ __launch_bounds__(256) void k_h(const float* __restrict__ x,
                                           const float* __restrict__ Ww,
                                           const float* __restrict__ Wb,
                                           float* __restrict__ hf32,
                                           _Float16* __restrict__ ht16)
{
    const int lane = threadIdx.x & 31;
    const int wave = threadIdx.x >> 5;
    const int row  = blockIdx.x * 8 + wave;          // [0, B*V*T)

    const float xv = x[(size_t)row * F0_ + lane];

    float w[32];
    const float4* Wv = reinterpret_cast<const float4*>(Ww + lane * F0_);
#pragma unroll
    for (int j = 0; j < 8; ++j) {
        float4 t4 = Wv[j];
        w[4 * j + 0] = t4.x; w[4 * j + 1] = t4.y;
        w[4 * j + 2] = t4.z; w[4 * j + 3] = t4.w;
    }
    float acc = Wb[lane];
#pragma unroll
    for (int f = 0; f < 32; ++f) acc = fmaf(__shfl(xv, f, 32), w[f], acc);
    const float h = lrelu(acc);

    hf32[(size_t)row * F1_ + lane] = h;

    const int t = row & 7;
    const int v = (row >> 3) & (V_ - 1);
    const int b = row >> 16;                          // row / (V*T)
    const int col = t * F1_ + lane;
    ht16[((size_t)b * COLS + col) * V_ + v] = (_Float16)h;
}

// ---------------------------------------------------------------------------
// Kernel 2: ht[v,g] = mean_b( sum_t h[b,v,t,g]*Wt[t] );
//           s_src[v] = ht·a[:32]; s_dst[v] = ht·a[32:].  One wave per v.
// ---------------------------------------------------------------------------
__global__ __launch_bounds__(256) void k_ht(const float* __restrict__ hf32,
                                            const float* __restrict__ Wt,
                                            const float* __restrict__ aw,
                                            float* __restrict__ s_src,
                                            float* __restrict__ s_dst)
{
    const int lane = threadIdx.x & 31;
    const int wave = threadIdx.x >> 5;
    const int v = blockIdx.x * 8 + wave;

    float ht = 0.f;
#pragma unroll
    for (int b = 0; b < B_; ++b)
#pragma unroll
        for (int t = 0; t < T_; ++t)
            ht = fmaf(hf32[(((size_t)(b * V_ + v)) * T_ + t) * F1_ + lane], Wt[t], ht);
    ht *= (1.0f / B_);

    float s0 = ht * aw[lane];
    float s1 = ht * aw[F1_ + lane];
#pragma unroll
    for (int off = 16; off > 0; off >>= 1) {
        s0 += __shfl_xor(s0, off, 32);
        s1 += __shfl_xor(s1, off, 32);
    }
    if (lane == 0) { s_src[v] = s0; s_dst[v] = s1; }
}

// ---------------------------------------------------------------------------
// Kernel 3: per-row masked-softmax stats (online max/sum) — one adj pass,
//           int4/float4 vectorized streaming.
// ---------------------------------------------------------------------------
__global__ __launch_bounds__(256) void k_stats(const int* __restrict__ adj,
                                               const float* __restrict__ s_src,
                                               const float* __restrict__ s_dst,
                                               float* __restrict__ mrow,
                                               float* __restrict__ zrow)
{
    __shared__ float sm[256], sz[256];
    const int i = blockIdx.x;
    const int tid = threadIdx.x;
    const float sd = s_dst[i];

    float m = -3.0e38f, z = 0.f;
    for (int it = 0; it < V_ / 1024; ++it) {
        const int j = it * 1024 + tid * 4;
        const int4   a4 = *reinterpret_cast<const int4*>(adj + (size_t)i * V_ + j);
        const float4 s4 = *reinterpret_cast<const float4*>(s_src + j);
        const int   av[4] = { a4.x, a4.y, a4.z, a4.w };
        const float sv[4] = { s4.x, s4.y, s4.z, s4.w };
#pragma unroll
        for (int u = 0; u < 4; ++u) {
            if (av[u] > 0) {
                float e = lrelu(sd + sv[u]);
                if (e > m) { z = z * __expf(m - e) + 1.f; m = e; }
                else       { z += __expf(e - m); }
            }
        }
    }
    sm[tid] = m; sz[tid] = z;
    __syncthreads();
    for (int s = 128; s > 0; s >>= 1) {
        if (tid < s) {
            float m1 = sm[tid], z1 = sz[tid];
            float m2 = sm[tid + s], z2 = sz[tid + s];
            float mm = fmaxf(m1, m2);
            sm[tid] = mm;
            sz[tid] = z1 * __expf(m1 - mm) + z2 * __expf(m2 - mm);
        }
        __syncthreads();
    }
    if (tid == 0) { mrow[i] = sm[0]; zrow[i] = sz[0]; }
}

// ---------------------------------------------------------------------------
// Kernel 4: fused attn-tile regeneration + WMMA GEMM.
//   block = 32 output rows (two M tiles) of one batch, all 256 columns.
//   Per 32-j chunk: 256 threads build a 32x32 f16 P tile (branchless,
//   int4/float4 loads, 8B LDS store) in double-buffered LDS; each of the
//   8 waves owns 2 N-tiles and both M-tiles -> 4 v_wmma_f32_16x16x32_f16
//   per chunk, with B fragments shared across the two M tiles.
// ---------------------------------------------------------------------------
__global__ __launch_bounds__(256) void k_attn_gemm(const int* __restrict__ adj,
                                                   const float* __restrict__ s_src,
                                                   const float* __restrict__ s_dst,
                                                   const float* __restrict__ mrow,
                                                   const float* __restrict__ zrow,
                                                   const _Float16* __restrict__ ht16,
                                                   float* __restrict__ out)
{
    __shared__ _Float16 P[2][32][32];

    const int tid  = threadIdx.x;
    const int lane = tid & 31;
    const int wv   = tid >> 5;                 // wave 0..7
    const int i0   = blockIdx.x * 32;
    const int b    = blockIdx.y;

    // P-production: thread fills row il, cols k0..k0+3
    const int   il   = tid >> 3;               // 0..31
    const int   irow = i0 + il;
    const float sd   = s_dst[irow];
    const float mr   = mrow[irow];
    const float zr   = zrow[irow];
    const float inz  = zr > 0.f ? 1.0f / zr : 0.f;
    const int   k0   = (tid & 7) * 4;
    const int*  adjrow = adj + (size_t)irow * V_;

    // WMMA fragment geometry (wave32)
    const int Ml = lane & 15;
    const int hi = lane >> 4;
    const int n0 = wv * 2;                     // first of this wave's 2 N-tiles
    const size_t hbase0 = ((size_t)b * COLS + (n0 * 16 + Ml)) * V_;
    const size_t hbase1 = ((size_t)b * COLS + ((n0 + 1) * 16 + Ml)) * V_;

    v8f c00 = {}, c01 = {}, c10 = {}, c11 = {};

    for (int chunk = 0; chunk < V_ / 32; ++chunk) {
        const int j0  = chunk * 32;
        const int buf = chunk & 1;

        // --- produce P tile (32x32 f16), branchless ---
        const int4   a4 = *reinterpret_cast<const int4*>(adjrow + j0 + k0);
        const float4 s4 = *reinterpret_cast<const float4*>(s_src + j0 + k0);
        __builtin_prefetch(adjrow + ((j0 + 128) & (V_ - 1)), 0, 1);
        float e0 = lrelu(sd + s4.x), e1 = lrelu(sd + s4.y);
        float e2 = lrelu(sd + s4.z), e3 = lrelu(sd + s4.w);
        float p0 = __expf(e0 - mr) * inz;  p0 = (a4.x > 0) ? p0 : 0.f;
        float p1 = __expf(e1 - mr) * inz;  p1 = (a4.y > 0) ? p1 : 0.f;
        float p2 = __expf(e2 - mr) * inz;  p2 = (a4.z > 0) ? p2 : 0.f;
        float p3 = __expf(e3 - mr) * inz;  p3 = (a4.w > 0) ? p3 : 0.f;
        union { _Float16 h[4]; uint2 u; } pk;
        pk.h[0] = (_Float16)p0; pk.h[1] = (_Float16)p1;
        pk.h[2] = (_Float16)p2; pk.h[3] = (_Float16)p3;
        *reinterpret_cast<uint2*>(&P[buf][il][k0]) = pk.u;
        __syncthreads();

        // --- A fragments (two 16x32 f16 M-tiles) from LDS ---
        union { uint4 q[2]; v16h h; } a0, a1;
        const _Float16* pb0 = &P[buf][Ml][0];
        const _Float16* pb1 = &P[buf][16 + Ml][0];
        a0.q[0] = *reinterpret_cast<const uint4*>(pb0 + hi * 8);
        a0.q[1] = *reinterpret_cast<const uint4*>(pb0 + 16 + hi * 8);
        a1.q[0] = *reinterpret_cast<const uint4*>(pb1 + hi * 8);
        a1.q[1] = *reinterpret_cast<const uint4*>(pb1 + 16 + hi * 8);

        // --- B fragments (32x16 f16) from pre-transposed H: contiguous 2x16B ---
        union { uint4 q[2]; v16h h; } b0, b1;
        const _Float16* hp0 = ht16 + hbase0 + j0 + hi * 8;
        const _Float16* hp1 = ht16 + hbase1 + j0 + hi * 8;
        b0.q[0] = *reinterpret_cast<const uint4*>(hp0);
        b0.q[1] = *reinterpret_cast<const uint4*>(hp0 + 16);
        b1.q[0] = *reinterpret_cast<const uint4*>(hp1);
        b1.q[1] = *reinterpret_cast<const uint4*>(hp1 + 16);

        c00 = __builtin_amdgcn_wmma_f32_16x16x32_f16(false, a0.h, false, b0.h,
                                                     (short)0, c00, false, false);
        c01 = __builtin_amdgcn_wmma_f32_16x16x32_f16(false, a0.h, false, b1.h,
                                                     (short)0, c01, false, false);
        c10 = __builtin_amdgcn_wmma_f32_16x16x32_f16(false, a1.h, false, b0.h,
                                                     (short)0, c10, false, false);
        c11 = __builtin_amdgcn_wmma_f32_16x16x32_f16(false, a1.h, false, b1.h,
                                                     (short)0, c11, false, false);
    }

    // epilogue: lrelu + store into (b, i, t, g) == (b, i, col)
#pragma unroll
    for (int r = 0; r < 8; ++r) {
        const int M  = r + hi * 8;
        const int N  = lane & 15;
        const size_t o0 = ((size_t)b * V_ + (i0 + M)) * COLS + N;
        const size_t o1 = ((size_t)b * V_ + (i0 + 16 + M)) * COLS + N;
        out[o0 + (size_t)n0 * 16]       = lrelu(c00[r]);
        out[o0 + (size_t)(n0 + 1) * 16] = lrelu(c01[r]);
        out[o1 + (size_t)n0 * 16]       = lrelu(c10[r]);
        out[o1 + (size_t)(n0 + 1) * 16] = lrelu(c11[r]);
    }
}

// ---------------------------------------------------------------------------
extern "C" void kernel_launch(void* const* d_in, const int* in_sizes, int n_in,
                              void* d_out, int out_size, void* d_ws, size_t ws_size,
                              hipStream_t stream)
{
    const float* x   = (const float*)d_in[0];
    const float* Ww  = (const float*)d_in[1];
    const float* Wb  = (const float*)d_in[2];
    const float* Wt  = (const float*)d_in[3];
    const float* aw  = (const float*)d_in[4];
    const int*   adj = (const int*)d_in[5];
    float* out = (float*)d_out;

    char* ws = (char*)d_ws;
    _Float16* ht16 = (_Float16*)(ws);                             //  8 MiB: Ht[b][col][j]
    float*    hf32 = (float*)(ws + (size_t)8  * 1024 * 1024);     // 16 MiB: h f32
    float*    ssrc = (float*)(ws + (size_t)24 * 1024 * 1024);     // 32 KiB
    float*    sdst = ssrc + V_;
    float*    mr   = sdst + V_;
    float*    zr   = mr + V_;

    k_h        <<<dim3((B_ * V_ * T_) / 8), 256, 0, stream>>>(x, Ww, Wb, hf32, ht16);
    k_ht       <<<dim3(V_ / 8),             256, 0, stream>>>(hf32, Wt, aw, ssrc, sdst);
    k_stats    <<<dim3(V_),                 256, 0, stream>>>(adj, ssrc, sdst, mr, zr);
    k_attn_gemm<<<dim3(V_ / 32, B_),        256, 0, stream>>>(adj, ssrc, sdst, mr, zr, ht16, out);
}
